// UPT_73632919323137
// MI455X (gfx1250) — compile-verified, matrix operand order checked by reading, not run on hardware
//
#include <hip/hip_runtime.h>

#define BB 16
#define NN 900
#define PP 4096
#define KK 117
#define GG 32
#define NMS_THR 0.5f
#define SCORE_THR 0.2f
#define FG_THR 0.5f

typedef __attribute__((ext_vector_type(16))) _Float16 v16h;
typedef __attribute__((ext_vector_type(8)))  float    v8f;

__device__ __forceinline__ float iou1(float ax1, float ay1, float ax2, float ay2,
                                      float bx1, float by1, float bx2, float by2) {
    float aa = (ax2 - ax1) * (ay2 - ay1);
    float ab = (bx2 - bx1) * (by2 - by1);
    float ix = fminf(ax2, bx2) - fmaxf(ax1, bx1);
    float iy = fminf(ay2, by2) - fmaxf(ay1, by1);
    ix = fmaxf(ix, 0.f);
    iy = fmaxf(iy, 0.f);
    float inter = ix * iy;
    return inter / (aa + ab - inter);
}

// ---------------- Kernel 1: class-aware greedy NMS, one block per batch ----
__global__ __launch_bounds__(256) void k_nms(const float* __restrict__ boxes,
                                             const float* __restrict__ scores,
                                             const int* __restrict__ labels,
                                             float* __restrict__ keepf) {
    __shared__ float sx1[NN], sy1[NN], sx2[NN], sy2[NN], sar[NN], ssc[NN];
    __shared__ int   sorder[NN], skept[NN];
    __shared__ float sred[256];
    __shared__ int   sflag;
    const int b = blockIdx.x, t = threadIdx.x, nt = blockDim.x;
    const float* bb = boxes + (size_t)b * NN * 4;

    // global max over this batch's box coords (for the class-offset trick)
    float m = -1e30f;
    for (int i = t; i < NN * 4; i += nt) m = fmaxf(m, bb[i]);
    sred[t] = m;
    __syncthreads();
    for (int s = 128; s > 0; s >>= 1) {
        if (t < s) sred[t] = fmaxf(sred[t], sred[t + s]);
        __syncthreads();
    }
    const float offmul = sred[0] + 1.0f;
    __syncthreads();

    for (int i = t; i < NN; i += nt) {
        float off = (float)labels[b * NN + i] * offmul;
        float x1 = bb[i * 4 + 0] + off, y1 = bb[i * 4 + 1] + off;
        float x2 = bb[i * 4 + 2] + off, y2 = bb[i * 4 + 3] + off;
        sx1[i] = x1; sy1[i] = y1; sx2[i] = x2; sy2[i] = y2;
        sar[i] = (x2 - x1) * (y2 - y1);
        ssc[i] = scores[b * NN + i];
        skept[i] = 0;
    }
    __syncthreads();

    // stable descending rank == jnp.argsort(-scores)
    for (int i = t; i < NN; i += nt) {
        float si = ssc[i];
        int r = 0;
        for (int j = 0; j < NN; j++) {
            float sj = ssc[j];
            r += (sj > si) || ((sj == si) && (j < i));
        }
        sorder[r] = i;
    }
    __syncthreads();

    // greedy scan over sorted positions
    for (int i = 0; i < NN; i++) {
        if (t == 0) sflag = 0;
        __syncthreads();
        int oi = sorder[i];
        float x1 = sx1[oi], y1 = sy1[oi], x2 = sx2[oi], y2 = sy2[oi], ai = sar[oi];
        int any = 0;
        for (int pos = t; pos < i; pos += nt) {
            if (skept[pos]) {
                int oj = sorder[pos];
                float ix = fminf(x2, sx2[oj]) - fmaxf(x1, sx1[oj]);
                float iy = fminf(y2, sy2[oj]) - fmaxf(y1, sy1[oj]);
                ix = fmaxf(ix, 0.f);
                iy = fmaxf(iy, 0.f);
                float inter = ix * iy;
                if (inter / (ai + sar[oj] - inter) > NMS_THR) any = 1;
            }
        }
        if (any) sflag = 1;
        __syncthreads();
        if (t == 0) skept[i] = (sflag == 0);
        __syncthreads();
    }

    for (int pos = t; pos < NN; pos += nt) {
        int orig = sorder[pos];
        keepf[b * NN + orig] =
            (skept[pos] && (ssc[orig] >= SCORE_THR)) ? 1.0f : 0.0f;
    }
}

// ---------------- Kernel 2: fg matrix (f16 {0,1}) + keep gate -------------
__global__ __launch_bounds__(256) void k_fg(const float* __restrict__ boxes,
                                            const float* __restrict__ keepf,
                                            const int* __restrict__ bh,
                                            const int* __restrict__ bo,
                                            const float* __restrict__ gth,
                                            const float* __restrict__ gto,
                                            float* __restrict__ gate,
                                            _Float16* __restrict__ fgA) {
    int gid = blockIdx.x * blockDim.x + threadIdx.x;
    if (gid >= BB * PP) return;
    int b = gid / PP;
    int ih = bh[gid], io = bo[gid];
    gate[gid] = keepf[b * NN + ih] * keepf[b * NN + io];
    const float* hb = boxes + ((size_t)b * NN + ih) * 4;
    const float* ob = boxes + ((size_t)b * NN + io) * 4;
    float hx1 = hb[0], hy1 = hb[1], hx2 = hb[2], hy2 = hb[3];
    float ox1 = ob[0], oy1 = ob[1], ox2 = ob[2], oy2 = ob[3];
    _Float16* dst = fgA + (size_t)gid * GG;
    for (int g = 0; g < GG; g++) {
        const float* gh = gth + ((size_t)b * GG + g) * 4;
        const float* go = gto + ((size_t)b * GG + g) * 4;
        float vh = iou1(hx1, hy1, hx2, hy2, gh[0], gh[1], gh[2], gh[3]);
        float vo = iou1(ox1, oy1, ox2, oy2, go[0], go[1], go[2], go[3]);
        dst[g] = (fminf(vh, vo) >= FG_THR) ? (_Float16)1.0f : (_Float16)0.0f;
    }
}

// ------- Kernel 3: WMMA einsum (fg x onehot) fused with focal loss --------
// block = 256 threads = 8 waves; block -> (batch, 16-row M tile);
// wave w -> class tile [16w, 16w+16). One v_wmma_f32_16x16x32_f16 per wave
// covers the full K-depth (G=32).
__global__ __launch_bounds__(256) void k_wmma_focal(const _Float16* __restrict__ fgA,
                                                    const float* __restrict__ gate,
                                                    const int* __restrict__ gt_labels,
                                                    const float* __restrict__ logits,
                                                    const float* __restrict__ prior,
                                                    float* __restrict__ accum) {
    __shared__ int sgtl[GG];
    const int b = blockIdx.x / (PP / 16);
    const int mtile = blockIdx.x % (PP / 16);
    const int t = threadIdx.x;
    if (t < GG) sgtl[t] = gt_labels[b * GG + t];
    __syncthreads();

    const int wave = t >> 5;       // N tile (8 tiles cover 128 >= K=117)
    const int lane = t & 31;
    const int hi   = lane >> 4;    // lane group 0/1
    const int mrow = lane & 15;    // row for A, column for B/C/D
    const int p0   = mtile * 16;

    // A: 16x32 f16, ISA layout -> 8 aligned dword loads per lane
    union { v16h v; unsigned u[8]; } A;
    const unsigned* ap =
        (const unsigned*)(fgA + ((size_t)b * PP + p0 + mrow) * GG);
#pragma unroll
    for (int v = 0; v < 8; v++) {
        int kb = ((v < 4) ? 2 * v : 2 * v + 8) + hi * 8;  // K index of low half
        A.u[v] = ap[kb >> 1];
    }

    // B: 32x16 one-hot of gt_labels, synthesized in registers
    const int kclass = wave * 16 + mrow;
    v16h Bm;
#pragma unroll
    for (int v = 0; v < 8; v++) {
        int kb = ((v < 4) ? 2 * v : 2 * v + 8) + hi * 8;
        Bm[2 * v]     = (sgtl[kb]     == kclass) ? (_Float16)1.0f : (_Float16)0.0f;
        Bm[2 * v + 1] = (sgtl[kb + 1] == kclass) ? (_Float16)1.0f : (_Float16)0.0f;
    }

    v8f C = {};
    C = __builtin_amdgcn_wmma_f32_16x16x32_f16(false, A.v, false, Bm,
                                               (short)0, C, false, false);

    float lsum = 0.f, npsum = 0.f;
    if (kclass < KK) {
#pragma unroll
        for (int r = 0; r < 8; r++) {
            int M = r + hi * 8;
            int p = p0 + M;
            float y = (C[r] > 0.f) ? 1.f : 0.f;   // labels = (einsum > 0)
            float g = gate[b * PP + p];
            size_t i0 = (((size_t)b * 2 + 0) * PP + p) * KK + kclass;
            size_t i1 = (((size_t)b * 2 + 1) * PP + p) * KK + kclass;
            float pr = prior[i0] * prior[i1] * g;
            if (pr > 0.f) {
                float tg = logits[((size_t)b * PP + p) * KK + kclass];
                float x = logf(pr / (1.f + expf(-tg) - pr) + 1e-8f);
                float sp = 1.f / (1.f + expf(-x));
                float pt = sp * y + (1.f - sp) * (1.f - y);
                float w = 0.5f;  // ALPHA = 0.5 -> both branches 0.5
                float lse = log1pf(expf(-fabsf(x)));
                float sp_pos = fmaxf(x, 0.f) + lse;    // softplus(x)
                float sp_neg = fmaxf(-x, 0.f) + lse;   // softplus(-x)
                float bce = sp_pos * (1.f - y) + sp_neg * y;
                float om = 1.f - pt;
                lsum += w * om * om * bce;             // GAMMA = 2
                npsum += y;
            }
        }
    }

    // wave32 reduction
#pragma unroll
    for (int off = 16; off > 0; off >>= 1) {
        lsum += __shfl_down(lsum, off);
        npsum += __shfl_down(npsum, off);
    }
    if (lane == 0) {
        atomicAdd(&accum[0], lsum);
        atomicAdd(&accum[1], npsum);
    }
}

__global__ void k_zero(float* accum) { accum[threadIdx.x] = 0.f; }

__global__ void k_final(const float* __restrict__ accum, float* __restrict__ out) {
    float np = accum[1];
    out[0] = accum[0] / ((np == 0.f) ? 1.f : np);
}

extern "C" void kernel_launch(void* const* d_in, const int* in_sizes, int n_in,
                              void* d_out, int out_size, void* d_ws, size_t ws_size,
                              hipStream_t stream) {
    const float* det_boxes   = (const float*)d_in[0];
    const float* det_scores  = (const float*)d_in[1];
    const float* pair_logits = (const float*)d_in[2];
    const float* prior       = (const float*)d_in[3];
    const float* gth         = (const float*)d_in[4];
    const float* gto         = (const float*)d_in[5];
    const int*   det_labels  = (const int*)d_in[6];
    const int*   bh          = (const int*)d_in[7];
    const int*   bo          = (const int*)d_in[8];
    const int*   gt_labels   = (const int*)d_in[9];

    char* ws = (char*)d_ws;
    float*    accum = (float*)ws;                                   // 2 floats
    float*    keepf = (float*)(ws + 64);                            // B*N
    float*    gate  = (float*)(ws + 64 + (size_t)BB * NN * 4);      // B*P
    _Float16* fgA   = (_Float16*)(ws + 64 + (size_t)BB * NN * 4 +
                                  (size_t)BB * PP * 4);             // B*P*G f16

    k_zero<<<1, 2, 0, stream>>>(accum);
    k_nms<<<BB, 256, 0, stream>>>(det_boxes, det_scores, det_labels, keepf);
    k_fg<<<(BB * PP + 255) / 256, 256, 0, stream>>>(det_boxes, keepf, bh, bo,
                                                    gth, gto, gate, fgA);
    k_wmma_focal<<<BB * (PP / 16), 256, 0, stream>>>(fgA, gate, gt_labels,
                                                     pair_logits, prior, accum);
    k_final<<<1, 1, 0, stream>>>(accum, (float*)d_out);
}
